// ImageTokenization_14044543058048
// MI455X (gfx1250) — matrix operand order repaired and checked by reading, not compiled, not run on hardware
//
#include <hip/hip_runtime.h>
#include <hip/hip_bf16.h>
#include <math.h>

// ---------------------------------------------------------------------------
// ImageTokenization for MI455X (gfx1250, wave32, WMMA, async-to-LDS).
//   stage 1: prep_weights        — fp32 -> bf16 weight repacking
//   stage 2: conv_bn_gelu_wmma   — implicit-GEMM 3x3 conv + BN + GELU + mask
//                                  (writes fp32 patches AND bf16 feature mat)
//   stage 3: proj_gemm_wmma      — 8192x768x2304 bf16 WMMA GEMM, double-
//                                  buffered GLOBAL_LOAD_ASYNC_TO_LDS staging
// All matrix math on V_WMMA_F32_16X16X32_BF16 (fp32 accumulate).
// ---------------------------------------------------------------------------

typedef __attribute__((ext_vector_type(16))) __bf16 v16bf;
typedef __attribute__((ext_vector_type(8)))  float  v8f;

union Frag {
    uint4  q[2];   // two 16-byte chunks = 8 VGPRs
    v16bf  v;
};

__device__ __forceinline__ float gelu_exact(float x) {
    return 0.5f * x * (1.0f + erff(x * 0.70710678118654752440f));
}

// Async copy: 16 B per lane, global -> LDS, tracked by ASYNCcnt.
// lds_off is the byte offset within the workgroup LDS allocation (low 32 bits
// of the generic pointer to a __shared__ object).
__device__ __forceinline__ void async_copy_b128(uint32_t lds_off, const void* gsrc) {
    asm volatile("global_load_async_to_lds_b128 %0, %1, off"
                 :: "v"(lds_off), "v"(gsrc)
                 : "memory");
}
__device__ __forceinline__ void wait_async_le3() {
    asm volatile("s_wait_asynccnt 0x3" ::: "memory");
}
__device__ __forceinline__ void wait_async_le0() {
    asm volatile("s_wait_asynccnt 0x0" ::: "memory");
}
__device__ __forceinline__ uint32_t lds_offset_of(const void* p) {
    return (uint32_t)(uintptr_t)p;   // LDS aperture: addr[31:0] = LDS offset
}

// ---------------------------------------------------------------------------
// Kernel 0: weight repack to bf16.
//   wconv[cout][pos][ci]  (pos = kh*3+kw, ci innermost so a 32-wide K step is
//                          32 contiguous input channels of one kernel tap)
//   wproj[d][f]           (straight convert; f = c*36 + py*6 + px already)
// ---------------------------------------------------------------------------
__global__ void prep_weights(const float* __restrict__ conv_w,
                             const float* __restrict__ proj_w,
                             __bf16* __restrict__ wconv,
                             __bf16* __restrict__ wproj) {
    int i = blockIdx.x * blockDim.x + threadIdx.x;
    if (i < 64 * 9 * 64) {
        int cout = i / 576;
        int rem  = i % 576;
        int pos  = rem / 64;
        int ci   = rem % 64;
        wconv[i] = (__bf16)conv_w[cout * 576 + ci * 9 + pos];
    }
    if (i < 768 * 2304) {
        wproj[i] = (__bf16)proj_w[i];
    }
}

// ---------------------------------------------------------------------------
// Kernel 1: implicit-GEMM conv 3x3 + bias + BN(eval) + exact GELU + mask.
// Grid: 32 frames * 96 rows. Block: 192 threads = 6 waves; wave w owns
// pixels [16w, 16w+16) of the row and all 64 output channels (4 WMMA accs).
// GEMM view per row: M=96 pixels, N=64 cout, K=576 (= 9 taps * 64 ci).
// LDS holds 3 padded input rows as [row 0..2][w 0..97][ci 0..63] bf16 so that
// an A-fragment K-chunk (8 consecutive ci) is one aligned ds_load_b128.
// Weight (B) fragments come straight from global: 72 KB, L2 resident, reused
// by all 3072 workgroups (prefetched at block start).
// ---------------------------------------------------------------------------
__global__ __launch_bounds__(192) void conv_bn_gelu_wmma(
    const float*  __restrict__ x,        // [32][64][96][96]
    const __bf16* __restrict__ wconv,    // [64][9*64]
    const float*  __restrict__ conv_b,
    const float*  __restrict__ gamma,
    const float*  __restrict__ beta,
    const float*  __restrict__ mean,
    const float*  __restrict__ var,
    const float*  __restrict__ mask,     // [6][6]
    float*        __restrict__ patches,  // [8192][64][6][6] fp32 (d_out part 2)
    __bf16*       __restrict__ afeat)    // [8192][2304] bf16 (stage-3 A matrix)
{
    __shared__ __align__(16) __bf16 sIn[3 * 98 * 64];   // 37632 B

    const int frame = blockIdx.x / 96;   // 0..31
    const int h     = blockIdx.x % 96;
    const int tid   = threadIdx.x;

    // warm the (hot, tiny) weight matrix into cache
    if (tid < 64) __builtin_prefetch(wconv + (size_t)tid * 576, 0, 1);

    // ---- stage 3 input rows (h-1, h, h+1) into LDS, transposed + padded ----
    {
        const int r  = tid / 64;         // 0..2
        const int ci = tid % 64;
        const int hh = h + r - 1;
        const bool inb = (hh >= 0) && (hh < 96);
        const float4* srow = (const float4*)(x + ((size_t)frame * 64 + ci) * 9216
                                               + (size_t)(inb ? hh : 0) * 96);
        const __bf16 zero = (__bf16)0.0f;
        sIn[(r * 98 + 0)  * 64 + ci] = zero;   // w = -1 padding
        sIn[(r * 98 + 97) * 64 + ci] = zero;   // w = 96 padding
        #pragma unroll
        for (int w4 = 0; w4 < 24; ++w4) {
            float4 f = inb ? srow[w4] : make_float4(0.f, 0.f, 0.f, 0.f);
            const int wb = w4 * 4 + 1;
            sIn[(r * 98 + wb + 0) * 64 + ci] = (__bf16)f.x;
            sIn[(r * 98 + wb + 1) * 64 + ci] = (__bf16)f.y;
            sIn[(r * 98 + wb + 2) * 64 + ci] = (__bf16)f.z;
            sIn[(r * 98 + wb + 3) * 64 + ci] = (__bf16)f.w;
        }
    }
    __syncthreads();

    const int wave = tid >> 5;       // 0..5  -> pixel group
    const int lane = tid & 31;
    const int lrow = lane & 15;      // A: M row / B: N col owned by this lane
    const int g    = lane >> 4;      // lane half (ISA K-chunk selector)

    v8f acc[4] = {v8f{}, v8f{}, v8f{}, v8f{}};

    // ---- K loop: 18 steps of 32 (one step = half the channels of one tap) --
    for (int ks = 0; ks < 18; ++ks) {
        const int pos = ks >> 1;            // kernel tap 0..8
        const int cib = (ks & 1) * 32;      // channel base within tap
        const int kh  = pos / 3;
        const int kw  = pos % 3;
        const int wc  = wave * 16 + lrow + kw;   // LDS w index (halo-shifted)

        Frag a;
        // A 16x32 bf16 layout: lane half g holds K-chunks {g*8..} and {16+g*8..}
        const uint4* ap = (const uint4*)&sIn[(kh * 98 + wc) * 64 + cib + g * 8];
        a.q[0] = ap[0];
        a.q[1] = ap[2];                     // +16 elements

        #pragma unroll
        for (int nt = 0; nt < 4; ++nt) {
            const int cout = nt * 16 + lrow;
            // B 32x16 bf16 layout: lane half g holds K-chunks {g*8..} and {16+g*8..}
            const uint4* bp = (const uint4*)(wconv + (size_t)cout * 576
                                                   + ks * 32 + g * 8);
            Frag b;
            b.q[0] = bp[0];
            b.q[1] = bp[2];
            acc[nt] = __builtin_amdgcn_wmma_f32_16x16x32_bf16(
                false, a.v, false, b.v, (short)0, acc[nt], false, false);
        }
    }

    // ---- fused epilogue: bias + BN + exact GELU + mask; dual store ---------
    const int py   = h % 6;
    const int prow = h / 6;
    #pragma unroll
    for (int nt = 0; nt < 4; ++nt) {
        const int cout = nt * 16 + lrow;
        const float sc = gamma[cout] * rsqrtf(var[cout] + 1e-5f);
        const float sh = (conv_b[cout] - mean[cout]) * sc + beta[cout];
        #pragma unroll
        for (int v = 0; v < 8; ++v) {
            const int w   = wave * 16 + v + 8 * g;   // C/D layout: M = v + 8*g
            float val = acc[nt][v] * sc + sh;
            val = gelu_exact(val);
            const int px = w % 6;
            val *= mask[py * 6 + px];
            const int    pidx = prow * 16 + (w / 6);
            const size_t row  = (size_t)frame * 256 + pidx;
            const size_t off  = row * 2304 + (size_t)cout * 36 + py * 6 + px;
            patches[off] = val;              // fp32 output #2
            afeat[off]   = (__bf16)val;      // bf16 A matrix for projection
        }
    }
}

// ---------------------------------------------------------------------------
// Kernel 2: token projection GEMM. M=8192, N=768, K=2304, bf16 in / fp32 acc.
// Block = 256 threads = 8 waves arranged 4(M) x 2(N): 64x128 output tile.
// Double-buffered async-to-LDS pipeline: per 32-wide K step a block stages
//   A tile  64x32 bf16 (4 KB)  -> 1 async_b128 issue per wave
//   B tile 128x32 bf16 (8 KB)  -> 2 async_b128 issues per wave
// The next stage is issued before draining the current one
// (s_wait_asynccnt 0x3), so HBM/L2 traffic overlaps the 4 WMMAs per wave.
// LDS: 2*(4+8) KB = 24 KB.
// ---------------------------------------------------------------------------
__global__ __launch_bounds__(256) void proj_gemm_wmma(
    const __bf16* __restrict__ A,        // [8192][2304]
    const __bf16* __restrict__ Bw,       // [768][2304]
    const float*  __restrict__ proj_b,   // [768]
    float*        __restrict__ tokens)   // [8192][768] (d_out part 1)
{
    __shared__ __align__(16) __bf16 sA[2][64 * 32];    // [buf][row][k]
    __shared__ __align__(16) __bf16 sB[2][128 * 32];   // [buf][col][k]

    const int mblk = blockIdx.x;         // 0..127
    const int nblk = blockIdx.y;         // 0..5
    const int tid  = threadIdx.x;
    const int wave = tid >> 5;
    const int lane = tid & 31;
    const int lrow = lane & 15;
    const int g    = lane >> 4;
    const int mw   = wave >> 1;          // 0..3
    const int nw   = wave & 1;           // 0..1

    const __bf16* Ablk = A  + (size_t)mblk * 64  * 2304;
    const __bf16* Bblk = Bw + (size_t)nblk * 128 * 2304;

    // this thread's staging chunks (8 bf16 = 16 B each)
    const int aRow  = tid >> 2;          // 0..63
    const int aKsub = (tid & 3) * 8;     // 0,8,16,24
    const int bCol0 = tid >> 1;          // chunk tid      -> cols 0..127
    const int bK0   = (tid & 1) * 16;    //                  ksub 0 or 16 (2 chunks)

    auto stage = [&](int kb, int buf) {
        // A: 256 chunks, one per thread
        async_copy_b128(lds_offset_of(&sA[buf][aRow * 32 + aKsub]),
                        Ablk + (size_t)aRow * 2304 + kb + aKsub);
        // B: 512 chunks, two per thread (32 contiguous B per chunk-pair)
        async_copy_b128(lds_offset_of(&sB[buf][bCol0 * 32 + bK0]),
                        Bblk + (size_t)bCol0 * 2304 + kb + bK0);
        async_copy_b128(lds_offset_of(&sB[buf][bCol0 * 32 + bK0 + 8]),
                        Bblk + (size_t)bCol0 * 2304 + kb + bK0 + 8);
    };

    v8f acc[4] = {v8f{}, v8f{}, v8f{}, v8f{}};

    stage(0, 0);                              // prologue
    for (int ks = 0; ks < 72; ++ks) {
        const int buf = ks & 1;
        if (ks < 71) {
            stage((ks + 1) * 32, buf ^ 1);    // keep next stage in flight
            wait_async_le3();                 // retire current stage only
        } else {
            wait_async_le0();
        }
        __syncthreads();                      // stage data visible block-wide

        Frag a;
        // A 16x32 layout: lane half g holds K-chunks {g*8..} and {16+g*8..}
        const uint4* ap = (const uint4*)&sA[buf][(mw * 16 + lrow) * 32 + g * 8];
        a.q[0] = ap[0];
        a.q[1] = ap[2];                       // +16 elements
        #pragma unroll
        for (int nt = 0; nt < 4; ++nt) {
            // B 32x16 layout: lane half g holds K-chunks {g*8..} and {16+g*8..}
            const uint4* bp =
                (const uint4*)&sB[buf][(nw * 64 + nt * 16 + lrow) * 32 + g * 8];
            Frag b;
            b.q[0] = bp[0];
            b.q[1] = bp[2];
            acc[nt] = __builtin_amdgcn_wmma_f32_16x16x32_bf16(
                false, a.v, false, b.v, (short)0, acc[nt], false, false);
        }
        __syncthreads();                      // protect buf before re-staging
    }

    #pragma unroll
    for (int nt = 0; nt < 4; ++nt) {
        const int col  = nblk * 128 + nw * 64 + nt * 16 + lrow;
        const float bb = proj_b[col];
        #pragma unroll
        for (int v = 0; v < 8; ++v) {
            const int m = mblk * 64 + mw * 16 + v + 8 * g;
            tokens[(size_t)m * 768 + col] = gelu_exact(acc[nt][v] + bb);
        }
    }
}

// ---------------------------------------------------------------------------
// Host launcher.
// Inputs (setup_inputs order): x, conv_w, conv_b, bn_gamma, bn_beta, bn_mean,
//                              bn_var, proj_w, proj_b, mask, patch_size
// d_out = tokens (8192*768 fp32) ++ patches (8192*2304 fp32)
// d_ws layout: wconv bf16 (73728 B) | wproj bf16 (3538944 B) | afeat bf16 (36 MB)
// ---------------------------------------------------------------------------
extern "C" void kernel_launch(void* const* d_in, const int* in_sizes, int n_in,
                              void* d_out, int out_size, void* d_ws, size_t ws_size,
                              hipStream_t stream) {
    const float* x      = (const float*)d_in[0];
    const float* conv_w = (const float*)d_in[1];
    const float* conv_b = (const float*)d_in[2];
    const float* gamma  = (const float*)d_in[3];
    const float* beta   = (const float*)d_in[4];
    const float* mean   = (const float*)d_in[5];
    const float* var    = (const float*)d_in[6];
    const float* proj_w = (const float*)d_in[7];
    const float* proj_b = (const float*)d_in[8];
    const float* mask   = (const float*)d_in[9];
    (void)in_sizes; (void)n_in; (void)out_size; (void)ws_size;

    char* ws = (char*)d_ws;
    __bf16* wconv = (__bf16*)(ws);
    __bf16* wproj = (__bf16*)(ws + 73728);
    __bf16* afeat = (__bf16*)(ws + 73728 + 3538944);

    float* tokens  = (float*)d_out;
    float* patches = (float*)d_out + (size_t)8192 * 768;

    prep_weights<<<(768 * 2304 + 255) / 256, 256, 0, stream>>>(
        conv_w, proj_w, wconv, wproj);

    conv_bn_gelu_wmma<<<32 * 96, 192, 0, stream>>>(
        x, wconv, conv_b, gamma, beta, mean, var, mask, patches, afeat);

    proj_gemm_wmma<<<dim3(128, 6), 256, 0, stream>>>(
        afeat, wproj, proj_b, tokens);
}
